// LSTM_37091337568423
// MI455X (gfx1250) — matrix operand order repaired and checked by reading, not compiled, not run on hardware
//
#include <hip/hip_runtime.h>
#include <hip/hip_bf16.h>

// ---------------------------------------------------------------------------
// Bidirectional 2-layer LSTM (H=32) on gfx1250 via v_wmma_f32_16x16x32_bf16.
//  - wave32: each wave owns a 16-row batch tile.
//  - gates g = [x|h] projection: 8 N-tiles (128 gate cols) of 16x16 f32 accum.
//  - Whh B-tiles resident in VGPRs; Wih B-tiles staged to LDS (bf16, B-layout).
//  - layer0: bias folded into Wih B-tile row k=4 against a constant 1.0 in the
//    A operand (K padding), so accumulators start from a persistent zero.
//  - h converted C-layout -> A-layout through a per-wave LDS buffer each step.
//  - fast activations: v_exp_f32 + raw v_rcp_f32 (no IEEE div chains).
//  - x(t+1) loaded RAW after the x(t) WMMAs are issued; converted at use ->
//    s_wait_loadcnt lands a full iteration after issue.
// ---------------------------------------------------------------------------

typedef __attribute__((ext_vector_type(16))) __bf16 v16bf;
typedef __attribute__((ext_vector_type(8)))  float  v8f;

#define HH   32
#define GG   128    // 4*H
#define TT   15     // steps per half
#define BB   32768  // original batch
#define BB2  65536  // batched halves

__device__ __forceinline__ float fast_rcp(float x) { return __builtin_amdgcn_rcpf(x); }
// sigmoid(x) = 1/(1+exp(-x)); rcp(inf)=0 -> saturates correctly, no clamp needed
__device__ __forceinline__ float sigm(float x)   { return fast_rcp(1.0f + __expf(-x)); }
// tanh(x) = 2*sigmoid(2x) - 1
__device__ __forceinline__ float tanh_f(float x) { return 2.0f * fast_rcp(1.0f + __expf(-2.0f * x)) - 1.0f; }

// DIN: input feature width (4 for layer0 from position_input, 64 for layer1
// from the previous layer's bf16 output). STORE_ALL: write every timestep
// (layer0) vs only the direction-final h (layer1).
template <int DIN, bool STORE_ALL>
__global__ __launch_bounds__(256)
void bilstm_layer_kernel(const void* __restrict__ x_f32_v,   // [BB,30,4] f32 (DIN==4)
                         const void* __restrict__ x_bf_v,    // [BB2,15,64] bf16 (DIN==64)
                         const float* __restrict__ Wih,      // [2,GG,DIN]
                         const float* __restrict__ Whh,      // [2,GG,HH]
                         const float* __restrict__ bias,     // [2,GG]
                         void* __restrict__ yout_v,          // [BB2,15,64] bf16
                         void* __restrict__ finals_v)        // [BB2,64] bf16
{
    constexpr int XCH = (DIN + 31) / 32;   // K-chunks of the input projection

    __shared__ v16bf sWihB[XCH * 8 * 32];                 // B-layout Wih (bf16)
    __shared__ __align__(16) __bf16 h_buf[8][16][32];     // per-wave h staging

    const float*  x_f32 = (const float*)x_f32_v;
    const __bf16* x_bf  = (const __bf16*)x_bf_v;
    __bf16* yout   = (__bf16*)yout_v;
    __bf16* finals = (__bf16*)finals_v;

    const int tid  = threadIdx.x;
    const int w    = tid >> 5;      // wave id in block
    const int L    = tid & 31;      // lane
    const int ln16 = L & 15;
    const bool hi  = (L >= 16);
    const int off8 = hi ? 8 : 0;
    const int dir  = blockIdx.y;    // 0 = fwd, 1 = bwd
    const int rowbase = blockIdx.x * 128 + w * 16;

    // ---- stage Wih into LDS in WMMA B-layout, zero-padded to K multiple of 32.
    // Operand o = c*8 + n covers gate cols [n*16, n*16+16), K = [c*32, c*32+32).
    // Layer0 (DIN==4): row k==DIN carries the combined bias (A has 1.0 at K=4).
    for (int v = tid; v < XCH * 8 * 32; v += 256) {
        int o  = v >> 5;
        int l2 = v & 31;
        int c  = o >> 3;
        int n  = o & 7;
        int g  = n * 16 + (l2 & 15);
        int kb = c * 32 + ((l2 >= 16) ? 8 : 0);
        const float* src = Wih + (size_t)dir * GG * DIN + (size_t)g * DIN;
        v16bf t;
#pragma unroll
        for (int i = 0; i < 16; ++i) {
            int k = kb + i + ((i >= 8) ? 8 : 0);
            float val = 0.0f;
            if (k < DIN) val = src[k];
            else if (DIN == 4 && k == DIN) val = bias[dir * GG + g];
            t[i] = (__bf16)val;
        }
        sWihB[v] = t;
    }
    __syncthreads();

    // ---- Whh resident in registers, B-layout: g[n] = sum_k h[k]*Whh[g][k],
    // so B[k][n] = Whh[n][k] -> lane reads a contiguous run of K from row g.
    v16bf whhB[8];
    {
        const int kb = hi ? 16 : 0;
#pragma unroll
        for (int n = 0; n < 8; ++n) {
            int g = n * 16 + ln16;
            const float* src = Whh + (size_t)dir * GG * HH + (size_t)g * HH + kb;
#pragma unroll
            for (int i = 0; i < 16; ++i) whhB[n][i] = (__bf16)src[i];
        }
    }

    // ---- layer1: combined bias splatted per step; layer0: folded into Wih
    float bb[8];
#pragma unroll
    for (int n = 0; n < 8; ++n) bb[n] = bias[dir * GG + n * 16 + ln16];

    // persistent zero accumulator seed (layer0 path)
    v8f vzero;
#pragma unroll
    for (int r = 0; r < 8; ++r) vzero[r] = 0.0f;

    // ---- x source base for this wave's 16 rows (row = rowbase + ln16)
    const int   brow  = rowbase + ln16;
    const int   srcb  = (brow < BB) ? brow : (brow - BB);
    const int   tb    = (brow < BB) ? 0 : TT;  // left / right half of T=30
    const float*  x0_base = x_f32 + (size_t)(srcb * 30 + tb) * 4;   // DIN==4
    const __bf16* x1_base = x_bf + (size_t)brow * TT * 64;          // DIN==64

    // ---- A operand state.
    // layer0: xa[0] elements 4..15 stay fixed (1.0 at K=4 for bias, rest 0);
    //         elements 0..3 are refreshed each step from the raw prefetch.
    // layer1: xa[] fully rewritten by the post-WMMA load each step.
    v16bf xa[XCH];
#pragma unroll
    for (int c = 0; c < XCH; ++c)
#pragma unroll
        for (int i = 0; i < 16; ++i) xa[c][i] = (__bf16)0.0f;
    if (DIN == 4 && !hi) xa[0][4] = (__bf16)1.0f;

    float4 xraw;                       // layer0 raw prefetch
    {
        const int t0 = dir ? (TT - 1) : 0;
        if constexpr (DIN == 4) {
            xraw = *(const float4*)(x0_base + (size_t)t0 * 4);
        } else {
            const __bf16* xr = x1_base + (size_t)t0 * 64;
#pragma unroll
            for (int c = 0; c < XCH; ++c)
#pragma unroll
                for (int i = 0; i < 8; ++i) {
                    xa[c][i]     = xr[c * 32 + off8 + i];
                    xa[c][8 + i] = xr[c * 32 + 16 + off8 + i];
                }
        }
    }

    // ---- cell state in C-layout registers (cols 0..31 -> 2 tiles)
    v8f cst[2];
#pragma unroll
    for (int j = 0; j < 2; ++j)
#pragma unroll
        for (int r = 0; r < 8; ++r) cst[j][r] = 0.0f;

#pragma unroll 1
    for (int t = 0; t < TT; ++t) {
        const int tt = dir ? (TT - 1 - t) : t;

        // layer0: convert this step's raw x into A elements 0..3 (waits here)
        if constexpr (DIN == 4) {
            if (!hi) {
                xa[0][0] = (__bf16)xraw.x;
                xa[0][1] = (__bf16)xraw.y;
                xa[0][2] = (__bf16)xraw.z;
                xa[0][3] = (__bf16)xraw.w;
            }
        }

        // accumulators: layer0 starts at 0 (bias is in Wih), layer1 at bias
        v8f acc[8];
#pragma unroll
        for (int n = 0; n < 8; ++n) {
            if (DIN == 4) {
                acc[n] = vzero;
            } else {
#pragma unroll
                for (int r = 0; r < 8; ++r) acc[n][r] = bb[n];
            }
        }

        // ---- input projection (Wih B-tiles streamed from LDS)
#pragma unroll
        for (int c = 0; c < XCH; ++c) {
#pragma unroll
            for (int n = 0; n < 8; ++n) {
                v16bf wb = sWihB[(c * 8 + n) * 32 + L];
                acc[n] = __builtin_amdgcn_wmma_f32_16x16x32_bf16(
                    false, xa[c], false, wb, (short)0, acc[n], false, false);
            }
        }

        // ---- issue next step's x load now that the WMMAs consumed xa
        if (t + 1 < TT) {
            const int ttn = dir ? (TT - 2 - t) : (t + 1);
            if constexpr (DIN == 4) {
                xraw = *(const float4*)(x0_base + (size_t)ttn * 4);
            } else {
                const __bf16* xr = x1_base + (size_t)ttn * 64;
#pragma unroll
                for (int c = 0; c < XCH; ++c)
#pragma unroll
                    for (int i = 0; i < 8; ++i) {
                        xa[c][i]     = xr[c * 32 + off8 + i];
                        xa[c][8 + i] = xr[c * 32 + 16 + off8 + i];
                    }
            }
        }

        // ---- recurrent projection (h == 0 at t == 0)
        if (t > 0) {
            v16bf ha;
            const __bf16* hr = &h_buf[w][ln16][0];
#pragma unroll
            for (int i = 0; i < 8; ++i) {
                ha[i]     = hr[2 * off8 + i];
                ha[8 + i] = hr[2 * off8 + 8 + i];
            }
#pragma unroll
            for (int n = 0; n < 8; ++n) {
                acc[n] = __builtin_amdgcn_wmma_f32_16x16x32_bf16(
                    false, ha, false, whhB[n], (short)0, acc[n], false, false);
            }
        }

        // ---- gates (PyTorch order i,f,g,o across tiles 0..7), state update,
        // write h back to LDS in row-major for next-step A-layout pickup.
#pragma unroll
        for (int j = 0; j < 2; ++j) {
#pragma unroll
            for (int r = 0; r < 8; ++r) {
                float iv = sigm(acc[0 + j][r]);
                float fv = sigm(acc[2 + j][r]);
                float gv = tanh_f(acc[4 + j][r]);
                float ov = sigm(acc[6 + j][r]);
                float cn = fv * cst[j][r] + iv * gv;
                cst[j][r] = cn;
                float hv = ov * tanh_f(cn);
                int M = r + (hi ? 8 : 0);           // C-layout row
                h_buf[w][M][j * 16 + ln16] = (__bf16)hv;
            }
        }
        __syncthreads();

        // ---- coalesced output: lanes 0..15 each flush one 32-col row (64B)
        if (STORE_ALL) {
            if (!hi) {
                const uint4* s = (const uint4*)&h_buf[w][ln16][0];
                uint4* d = (uint4*)(yout + ((size_t)(rowbase + ln16) * TT + tt) * 64
                                    + dir * 32);
#pragma unroll
                for (int q = 0; q < 4; ++q) d[q] = s[q];
            }
        } else if (t == TT - 1) {
            if (!hi) {
                const uint4* s = (const uint4*)&h_buf[w][ln16][0];
                uint4* d = (uint4*)(finals + (size_t)(rowbase + ln16) * 64 + dir * 32);
#pragma unroll
                for (int q = 0; q < 4; ++q) d[q] = s[q];
            }
        }
        __syncthreads();
    }
}

// ---------------------------------------------------------------------------
// Head: argmax(dir_input) masks + the two FC layers.
// finals layout: [model(2)][BB2][64] bf16; row b = left half, b+BB = right.
// Per model-row: cols 0..31 = fwd h(t=14), cols 32..63 = bwd h(t=0).
// ---------------------------------------------------------------------------
__global__ __launch_bounds__(256)
void lstm_head_kernel(const float* __restrict__ dir_input,   // [BB,6]
                      const void* __restrict__ finals_v,     // [2][BB2][64] bf16
                      const float* __restrict__ lvlW,        // [2,64]
                      const float* __restrict__ lvlb,        // [2]
                      const float* __restrict__ vorW,        // [2,128]
                      const float* __restrict__ vorb,        // [2]
                      float* __restrict__ out)               // [BB,2]
{
    int b = blockIdx.x * blockDim.x + threadIdx.x;
    if (b >= BB) return;

    const __bf16* finals = (const __bf16*)finals_v;

    const float* di = dir_input + (size_t)b * 6;
    int am = 0; float mx = di[0];
#pragma unroll
    for (int k = 1; k < 6; ++k) { float v = di[k]; if (v > mx) { mx = v; am = k; } }
    float m_vor = (am == 1 || am == 4) ? 1.0f : 0.0f;
    float m_r   = (am == 0 || am == 5) ? 1.0f : 0.0f;
    float m_l   = (am == 2 || am == 3) ? 1.0f : 0.0f;

    const __bf16* lvlL = finals + (size_t)b * 64;
    const __bf16* lvlR = finals + (size_t)(b + BB) * 64;
    const __bf16* vorL = finals + (size_t)BB2 * 64 + (size_t)b * 64;
    const __bf16* vorR = finals + (size_t)BB2 * 64 + (size_t)(b + BB) * 64;

#pragma unroll
    for (int o = 0; o < 2; ++o) {
        float aL = lvlb[o], aR = lvlb[o], aV = vorb[o];
#pragma unroll 4
        for (int c = 0; c < 64; ++c) {
            float wl = lvlW[o * 64 + c];
            aL += wl * (float)lvlL[c];
            aR += wl * (float)lvlR[c];
            aV += vorW[o * 128 + c]      * (float)vorL[c];
            aV += vorW[o * 128 + 64 + c] * (float)vorR[c];
        }
        out[(size_t)b * 2 + o] = aR * m_r + aL * m_l + aV * m_vor;
    }
}

// ---------------------------------------------------------------------------
extern "C" void kernel_launch(void* const* d_in, const int* in_sizes, int n_in,
                              void* d_out, int out_size, void* d_ws, size_t ws_size,
                              hipStream_t stream)
{
    (void)in_sizes; (void)n_in; (void)out_size; (void)ws_size;

    const float* dir_input = (const float*)d_in[0];
    const float* pos       = (const float*)d_in[1];

    // workspace: y0 = [BB2,15,64] bf16, finals = [2][BB2,64] bf16
    char*  ws      = (char*)d_ws;
    size_t y0Elems = (size_t)BB2 * TT * 64;
    void*  y0      = (void*)ws;
    char*  finals0 = ws + y0Elems * sizeof(unsigned short);
    size_t fStride = (size_t)BB2 * 64 * sizeof(unsigned short);

    dim3 grid(BB2 / 128, 2);   // 512 batch tiles x 2 directions
    dim3 block(256);

    for (int m = 0; m < 2; ++m) {               // m=0: lvl, m=1: vor
        int base = 2 + m * 6;
        const float* Wih0 = (const float*)d_in[base + 0];
        const float* Whh0 = (const float*)d_in[base + 1];
        const float* b0   = (const float*)d_in[base + 2];
        const float* Wih1 = (const float*)d_in[base + 3];
        const float* Whh1 = (const float*)d_in[base + 4];
        const float* b1   = (const float*)d_in[base + 5];
        void* finalsM = (void*)(finals0 + (size_t)m * fStride);

        bilstm_layer_kernel<4, true><<<grid, block, 0, stream>>>(
            (const void*)pos, nullptr, Wih0, Whh0, b0, y0, nullptr);
        bilstm_layer_kernel<64, false><<<grid, block, 0, stream>>>(
            nullptr, (const void*)y0, Wih1, Whh1, b1, nullptr, finalsM);
    }

    lstm_head_kernel<<<dim3(BB / 256), block, 0, stream>>>(
        dir_input, (const void*)finals0,
        (const float*)d_in[14], (const float*)d_in[15],
        (const float*)d_in[16], (const float*)d_in[17],
        (float*)d_out);
}
